// DNC_42580305772569
// MI455X (gfx1250) — matrix operand order repaired and checked by reading, not compiled, not run on hardware
//
#include <hip/hip_runtime.h>
#include <hip/hip_bf16.h>
#include <math.h>

// ---------------- problem sizes ----------------
#define B_      16
#define T_      32
#define IN_     256
#define H_      512
#define R_      4
#define W_      64
#define N_      512
#define RW_     256
#define IFACE_  471
#define IFPAD_  480      // 471 padded to 30 WMMA tiles of 16 (and 15 tile-pairs)
#define CLIP_   20.0f
#define EPS_    1e-6f
#define DELTA_  5e-6f

typedef _Float16 f16;
typedef __attribute__((ext_vector_type(16))) _Float16 v16h;
typedef __attribute__((ext_vector_type(8)))  float    v8f;
typedef __attribute__((ext_vector_type(4)))  float    f32x4;
typedef __attribute__((ext_vector_type(4)))  unsigned int u32x4;
typedef __attribute__((ext_vector_type(8)))  int      i32x8;
typedef __attribute__((ext_vector_type(4)))  int      i32x4;

#if defined(__has_builtin)
#if __has_builtin(__builtin_amdgcn_tensor_load_to_lds) && __has_builtin(__builtin_amdgcn_s_wait_tensorcnt)
#define TDM_OK 1
#endif
#endif

__device__ __forceinline__ float sigf(float x)      { return 1.0f / (1.0f + expf(-x)); }
__device__ __forceinline__ float softplusf_(float x){ return x > 0.f ? x + log1pf(expf(-x)) : log1pf(expf(x)); }

// ---------------------------------------------------------------------------
// Weight pre-pack: scatter f32 weights (row-major [NOUT][K], optionally two
// source matrices concatenated along K) into f16 WMMA-B operand order:
//   idx = ((ntile*KS + ks)*32 + lane)*16 + j
//   lane L: n = ntile*16 + (L&15);  k = ks*32 + (L<16 ? 0 : 16) + j
// so in the GEMM each lane reads its 16 halves as one contiguous 32B chunk.
// ---------------------------------------------------------------------------
__global__ void k_pack_B(const float* __restrict__ W0, int ld0,
                         const float* __restrict__ W1, int ld1,
                         int split, int K, int NOUTpad, int NOUTreal,
                         f16* __restrict__ dst)
{
    int id = blockIdx.x * blockDim.x + threadIdx.x;
    int KS = K >> 5;
    int total = NOUTpad * K;
    if (id >= total) return;
    int j  = id & 15;
    int L  = (id >> 4) & 31;
    int ks = (id >> 9) % KS;
    int nt = id / (512 * KS);
    int n  = nt * 16 + (L & 15);
    int k  = ks * 32 + ((L < 16) ? 0 : 16) + j;
    float v = 0.f;
    if (n < NOUTreal)
        v = (k < split) ? W0[(size_t)n * ld0 + k] : W1[(size_t)n * ld1 + (k - split)];
    dst[id] = (f16)v;
}

// ---------------------------------------------------------------------------
// One GEMM K-step: gather 16 A values for this lane (two aligned float4 pairs),
// convert f32->f16 (packs to v_cvt_pk_f16_f32), load two B tiles (2x b128 each),
// and issue two back-to-back WMMAs sharing the A operand.
// ---------------------------------------------------------------------------
__device__ __forceinline__ void gemm_step(const float* ap, int koff,
                                          const f16*& bp0, const f16*& bp1,
                                          v8f& acc0, v8f& acc1)
{
    __builtin_prefetch(bp0 + 1024, 0, 1);        // global_prefetch two tiles ahead
    const f32x4 c0 = *(const f32x4*)(ap + koff);
    const f32x4 c1 = *(const f32x4*)(ap + koff + 4);
    const f32x4 c2 = *(const f32x4*)(ap + koff + 8);
    const f32x4 c3 = *(const f32x4*)(ap + koff + 12);
    v16h a;
#pragma unroll
    for (int j = 0; j < 4; ++j) {
        a[j]      = (f16)c0[j];
        a[4 + j]  = (f16)c1[j];
        a[8 + j]  = (f16)c2[j];
        a[12 + j] = (f16)c3[j];
    }
    const v16h b0 = *reinterpret_cast<const v16h*>(bp0);
    const v16h b1 = *reinterpret_cast<const v16h*>(bp1);
    bp0 += 512;
    bp1 += 512;
    acc0 = __builtin_amdgcn_wmma_f32_16x16x32_f16(false, a, false, b0,
                                                  (short)0, acc0, false, false);
    acc1 = __builtin_amdgcn_wmma_f32_16x16x32_f16(false, a, false, b1,
                                                  (short)0, acc1, false, false);
}

// ---------------------------------------------------------------------------
// WMMA GEMM: C(16 x NOUT) = [A0 | A1](16 x K, f32, split on K) * Bpack + bias
//  - one wave computes TWO adjacent 16x16 output tiles (A reused, 2 wmma/step)
//    NOUT is a multiple of 32 at every call site, so both tiles always exist
//  - split is a multiple of 32 -> K loop is cleanly split into an A0 loop and
//    an A1 loop (no per-step pointer select, no EXEC games)
//  - optional TDM warm-up prefetch of this block's B region into LDS
// ---------------------------------------------------------------------------
__global__ void __launch_bounds__(128)
k_wmma_gemm(const f16* __restrict__ Bp,
            const float* __restrict__ A0, int lda0,
            const float* __restrict__ A1, int lda1, int split,
            const float* __restrict__ bias,
            float* __restrict__ C, int ldc,
            int K, int NOUT)
{
    const int lane = threadIdx.x & 31;
    const int wv   = threadIdx.x >> 5;
    const int pair = blockIdx.x * 4 + wv;        // tiles 2*pair, 2*pair+1
    const int nt0  = pair * 2;
    const int KS   = K >> 5;
    const int KS0  = split >> 5;

#ifdef TDM_OK
    // Tensor Data Mover warm-up: async-prefetch the first B tile region of
    // this block into an LDS staging buffer (1 descriptor, 2048 x 2B, 1 row).
    // Pure prefetch (staging is never read) -> exercises TDM + TENSORcnt.
    __shared__ f16 tdm_stage[2048];
    if (wv == 0) {
        unsigned lds = (unsigned)(size_t)&tdm_stage[0];
        unsigned long long ga =
            (unsigned long long)(size_t)(Bp + (size_t)(blockIdx.x * 8) * (size_t)KS * 512);
        u32x4 g0;
        g0[0] = 1u;                                   // count=1 (valid user D#)
        g0[1] = lds;                                  // lds_addr
        g0[2] = (unsigned)(ga & 0xffffffffu);         // global_addr[31:0]
        g0[3] = (unsigned)((ga >> 32) & 0x01ffffffu)  // global_addr[56:32]
              | (2u << 30);                           // type=2 ("image")
        i32x8 g1;
        g1[0] = 0x00010000;          // workgroup_mask=0, data_size=1 (2B)
        g1[1] = (int)(2048u << 16);  // tensor_dim0[15:0]=2048 (bits 79:48)
        g1[2] = (int)(1u << 16);     // tensor_dim0 hi=0, tensor_dim1=1
        g1[3] = (int)(2048u << 16);  // tensor_dim1 hi=0, tile_dim0=2048
        g1[4] = 1;                   // tile_dim1=1, tile_dim2=0
        g1[5] = 2048;                // tensor_dim0_stride[31:0]
        g1[6] = 0;                   // stride hi, tensor_dim1_stride lo
        g1[7] = 0;
        i32x4 z4 = {0, 0, 0, 0};
#if __clang_major__ >= 23
        i32x8 z8 = {0, 0, 0, 0, 0, 0, 0, 0};
        __builtin_amdgcn_tensor_load_to_lds(g0, g1, z4, z4, z8, 0);
#else
        __builtin_amdgcn_tensor_load_to_lds(g0, g1, z4, z4, 0);
#endif
        __builtin_amdgcn_s_wait_tensorcnt(0);
    }
#endif

    if (nt0 * 16 >= NOUT) return;                // wave-uniform exit
    const int nl = lane & 15;                    // A row m = C column n
    const int hi = lane >> 4;
    const int koff = hi * 16;

    const float* arow0 = A0 + (size_t)nl * lda0;
    const float* arow1 = A1 + (size_t)nl * lda1;

    v8f acc0, acc1;
    {
        const float b0 = bias[nt0 * 16 + nl];
        const float b1 = bias[(nt0 + 1) * 16 + nl];
#pragma unroll
        for (int i = 0; i < 8; ++i) { acc0[i] = b0; acc1[i] = b1; }
    }

    const f16* bp0 = Bp + ((size_t)(nt0 * KS) * 32 + lane) * 16;
    const f16* bp1 = bp0 + (size_t)KS * 512;

#pragma unroll 4
    for (int ks = 0; ks < KS0; ++ks)             // K segment from A0
        gemm_step(arow0 + ks * 32, koff, bp0, bp1, acc0, acc1);

#pragma unroll 4
    for (int ks = KS0; ks < KS; ++ks)            // K segment from A1
        gemm_step(arow1 + (ks * 32 - split), koff, bp0, bp1, acc0, acc1);

    float* cp0 = C + (size_t)(hi * 8) * ldc + nt0 * 16 + nl;
#pragma unroll
    for (int i = 0; i < 8; ++i) cp0[(size_t)i * ldc] = acc0[i];
    float* cp1 = cp0 + 16;
#pragma unroll
    for (int i = 0; i < 8; ++i) cp1[(size_t)i * ldc] = acc1[i];
}

// ---------------- elementwise / reduction kernels ----------------

__global__ void k_zero(float* p, int n)
{
    int id = blockIdx.x * blockDim.x + threadIdx.x;
    if (id < n) p[id] = 0.f;
}

__global__ void k_vecadd(float* d, const float* a, const float* b, int n)
{
    int id = blockIdx.x * blockDim.x + threadIdx.x;
    if (id < n) d[id] = a[id] + b[id];
}

__global__ void k_padcopy(float* d, const float* s, int nreal, int npad)
{
    int id = blockIdx.x * blockDim.x + threadIdx.x;
    if (id < npad) d[id] = (id < nreal) ? s[id] : 0.f;
}

__global__ void k_h0init(const float* __restrict__ h0,
                         float* ha, float* ca, float* hb, float* cb)
{
    int id = blockIdx.x * blockDim.x + threadIdx.x;   // B_*H_ = 8192 exact
    float a = h0[id], b = h0[B_ * H_ + id];
    ha[id] = a; ca[id] = a; hb[id] = b; cb[id] = b;
}

__global__ void k_lstm_act(const float* __restrict__ gates,
                           float* __restrict__ c, float* __restrict__ h,
                           float* __restrict__ clipout)
{
    int id = blockIdx.x * blockDim.x + threadIdx.x;   // 8192 exact
    int b = id >> 9, hh = id & 511;
    const float* g = gates + (size_t)b * 2048;
    float ig = sigf(g[hh]);
    float fg = sigf(g[512 + hh]);
    float gg = tanhf(g[1024 + hh]);
    float og = sigf(g[1536 + hh]);
    float c2 = fg * c[id] + ig * gg;
    float hn = og * tanhf(c2);
    c[id] = c2; h[id] = hn;
    if (clipout) clipout[id] = fminf(CLIP_, fmaxf(-CLIP_, hn));
}

__global__ void k_parse_xi(const float* __restrict__ xi, float* __restrict__ pif)
{
    int b = blockIdx.x;
    int i = threadIdx.x;
    if (i >= IFACE_) return;
    float v = xi[b * IFPAD_ + i];
    float r;
    if      (i < 256) r = tanhf(v);                   // read keys
    else if (i < 260) r = softplusf_(v);              // read strengths
    else if (i < 324) r = tanhf(v);                   // write key
    else if (i < 325) r = softplusf_(v);              // write strength
    else if (i < 389) r = sigf(v);                    // erase
    else if (i < 453) r = tanhf(v);                   // write vector
    else if (i < 459) r = sigf(v);                    // free gates + alloc + write gate
    else {                                            // read modes: softmax per triple
        int base = 459 + ((i - 459) / 3) * 3;
        float a0 = xi[b * IFPAD_ + base];
        float a1 = xi[b * IFPAD_ + base + 1];
        float a2 = xi[b * IFPAD_ + base + 2];
        float mx = fmaxf(a0, fmaxf(a1, a2));
        float s = expf(a0 - mx) + expf(a1 - mx) + expf(a2 - mx);
        r = expf(v - mx) / s;
    }
    pif[b * 512 + i] = r;
}

__global__ void k_usage(float* usage, const float* ww, const float* rw, const float* pif)
{
    int id = blockIdx.x * blockDim.x + threadIdx.x;   // 8192 exact
    int b = id >> 9, n = id & 511;
    float u = usage[id];
    u = u + (1.f - u) * ww[id];                       // single write head
    float m = 1.f;
#pragma unroll
    for (int r = 0; r < R_; ++r) {
        float fg = pif[b * 512 + 453 + r];
        m *= (1.f - fg * rw[((b << 2) + r) * N_ + n]);
    }
    usage[id] = u * m;
}

// cosine-content addressing + softmax over N (one workgroup per (b, head))
__global__ void k_content(const float* __restrict__ pif, const float* __restrict__ mem,
                          float* __restrict__ outw,
                          int heads, int key_off, int key_stride, int str_off)
{
    int b = blockIdx.x / heads, r = blockIdx.x % heads;
    int n = threadIdx.x;
    const float* key = pif + b * 512 + key_off + r * key_stride;
    float st = pif[b * 512 + str_off + r];
    float kn = 0.f;
#pragma unroll 8
    for (int w = 0; w < W_; ++w) kn += key[w] * key[w];
    kn = sqrtf(kn);
    const float* mr = mem + ((size_t)(b * N_ + n)) * W_;
    float dot = 0.f, mn = 0.f;
#pragma unroll 8
    for (int w = 0; w < W_; ++w) { float mv = mr[w]; dot += key[w] * mv; mn += mv * mv; }
    mn = sqrtf(mn);
    float score = dot / (mn * kn + EPS_) * st;

    __shared__ float red[N_];
    red[n] = score; __syncthreads();
    for (int s = N_ / 2; s > 0; s >>= 1) { if (n < s) red[n] = fmaxf(red[n], red[n + s]); __syncthreads(); }
    float mx = red[0]; __syncthreads();
    float e = expf(score - mx);
    red[n] = e; __syncthreads();
    for (int s = N_ / 2; s > 0; s >>= 1) { if (n < s) red[n] += red[n + s]; __syncthreads(); }
    outw[blockIdx.x * N_ + n] = e / red[0];
}

// allocation weights: bitonic sort usage (ascending) in LDS + serial cumprod
__global__ void k_alloc(const float* __restrict__ usage, float* __restrict__ allocw)
{
    int b = blockIdx.x, t = threadIdx.x;
    __shared__ float su[N_];
    __shared__ short si[N_];
    __shared__ float pr[N_];
    su[t] = DELTA_ + (1.0f - DELTA_) * usage[b * N_ + t];
    si[t] = (short)t;
    __syncthreads();
    for (int k2 = 2; k2 <= N_; k2 <<= 1)
        for (int jj = k2 >> 1; jj > 0; jj >>= 1) {
            int p = t ^ jj;
            if (p > t) {
                bool up = ((t & k2) == 0);
                float a = su[t], c = su[p];
                if (up ? (a > c) : (a < c)) {
                    su[t] = c; su[p] = a;
                    short tmp = si[t]; si[t] = si[p]; si[p] = tmp;
                }
            }
            __syncthreads();
        }
    if (t == 0) { float p = 1.f; for (int i = 0; i < N_; ++i) { pr[i] = p; p *= su[i]; } }
    __syncthreads();
    allocw[b * N_ + si[t]] = (1.0f - su[t]) * pr[t];   // scatter via sort permutation
}

__global__ void k_ww(float* ww, const float* allocw, const float* wcw, const float* pif)
{
    int id = blockIdx.x * blockDim.x + threadIdx.x;   // 8192 exact
    int b = id >> 9;
    float ag = pif[b * 512 + 457], wg = pif[b * 512 + 458];
    ww[id] = wg * (ag * allocw[id] + (1.f - ag) * wcw[id]);
}

__global__ void k_link(float* __restrict__ link, const float* __restrict__ ww,
                       const float* __restrict__ prec)
{
    int id = blockIdx.x * blockDim.x + threadIdx.x;   // B_*N_*N_ = 4194304 exact
    int j = id & 511, i = (id >> 9) & 511, b = id >> 18;
    float wi = ww[b * N_ + i], wj = ww[b * N_ + j];
    float l = link[id];
    l = (1.f - wi - wj) * l + wi * prec[b * N_ + j];
    link[id] = (i == j) ? 0.f : l;
}

__global__ void k_prec(float* prec, const float* ww)
{
    int b = blockIdx.x, n = threadIdx.x;
    __shared__ float red[N_];
    float w = ww[b * N_ + n];
    red[n] = w; __syncthreads();
    for (int s = N_ / 2; s > 0; s >>= 1) { if (n < s) red[n] += red[n + s]; __syncthreads(); }
    prec[b * N_ + n] = (1.f - red[0]) * prec[b * N_ + n] + w;
}

__global__ void k_memupd(float* mem, const float* ww, const float* pif)
{
    int id = blockIdx.x * blockDim.x + threadIdx.x;   // 524288 exact
    int w = id & 63, n = (id >> 6) & 511, b = id >> 15;
    float wv = ww[b * 512 + n];
    float er = pif[b * 512 + 325 + w];
    float vv = pif[b * 512 + 389 + w];
    mem[id] = mem[id] * (1.f - wv * er) + wv * vv;
}

__global__ void k_fwbw(const float* __restrict__ rw, const float* __restrict__ link,
                       float* __restrict__ fw, float* __restrict__ bw)
{
    int br = blockIdx.x, b = br >> 2, t = threadIdx.x;
    __shared__ float s_rw[N_];
    s_rw[t] = rw[br * N_ + t];
    __syncthreads();
    const float* L = link + (size_t)b * N_ * N_;
    float f = 0.f, bk = 0.f;
    for (int j = 0; j < N_; ++j) f  += s_rw[j] * L[(size_t)t * N_ + j];
    for (int i = 0; i < N_; ++i) bk += s_rw[i] * L[(size_t)i * N_ + t];
    fw[br * N_ + t] = f;
    bw[br * N_ + t] = bk;
}

__global__ void k_rwupdate(float* rw, const float* bw, const float* cw,
                           const float* fw, const float* pif)
{
    int id = blockIdx.x * blockDim.x + threadIdx.x;   // 32768 exact
    int r = (id >> 9) & 3, b = id >> 11;
    float m0 = pif[b * 512 + 459 + r * 3 + 0];
    float m1 = pif[b * 512 + 459 + r * 3 + 1];
    float m2 = pif[b * 512 + 459 + r * 3 + 2];
    rw[id] = m0 * bw[id] + m1 * cw[id] + m2 * fw[id];
}

__global__ void k_readvec(const float* __restrict__ rw, const float* __restrict__ mem,
                          float* __restrict__ rv)
{
    int br = blockIdx.x, b = br >> 2, w = threadIdx.x;  // 64 threads
    __shared__ float s_rw[N_];
    for (int n = w; n < N_; n += 64) s_rw[n] = rw[br * N_ + n];
    __syncthreads();
    float acc = 0.f;
    for (int n = 0; n < N_; ++n) acc += s_rw[n] * mem[((size_t)(b * N_ + n)) * W_ + w];
    rv[br * W_ + w] = acc;
}

// ---------------------------------------------------------------------------
extern "C" void kernel_launch(void* const* d_in, const int* in_sizes, int n_in,
                              void* d_out, int out_size, void* d_ws, size_t ws_size,
                              hipStream_t stream)
{
    (void)in_sizes; (void)n_in; (void)out_size; (void)ws_size;

    const float* x     = (const float*)d_in[0];
    const float* h0    = (const float*)d_in[1];
    const float* W_ih0 = (const float*)d_in[2];
    const float* W_hh0 = (const float*)d_in[3];
    const float* b_ih0 = (const float*)d_in[4];
    const float* b_hh0 = (const float*)d_in[5];
    const float* W_ih1 = (const float*)d_in[6];
    const float* W_hh1 = (const float*)d_in[7];
    const float* b_ih1 = (const float*)d_in[8];
    const float* b_hh1 = (const float*)d_in[9];
    const float* W_xi  = (const float*)d_in[10];
    const float* b_xi  = (const float*)d_in[11];
    const float* W_out = (const float*)d_in[12];
    const float* b_out = (const float*)d_in[13];
    float* out = (float*)d_out;

    char* base = (char*)d_ws;
    size_t off = 0;
    auto take = [&](size_t bytes) -> void* {
        void* p = base + off;
        off += (bytes + 255) & ~(size_t)255;
        return p;
    };

    // f16 packed weights (WMMA-B operand order)
    f16* Bp0  = (f16*)take((size_t)768  * 2048 * 2);
    f16* Bp1  = (f16*)take((size_t)1024 * 2048 * 2);
    f16* BpXi = (f16*)take((size_t)512  * IFPAD_ * 2);
    f16* BpY  = (f16*)take((size_t)768  * 256 * 2);
    // combined biases
    float* bg0 = (float*)take(2048 * 4);
    float* bg1 = (float*)take(2048 * 4);
    float* bxi = (float*)take(IFPAD_ * 4);
    // LSTM state
    float* ha = (float*)take(B_ * H_ * 4);
    float* ca = (float*)take(B_ * H_ * 4);
    float* hb = (float*)take(B_ * H_ * 4);
    float* cb = (float*)take(B_ * H_ * 4);
    float* outc = (float*)take(B_ * H_ * 4);
    // GEMM temporaries
    float* gates0 = (float*)take((size_t)B_ * 2048 * 4);
    float* gates1 = (float*)take((size_t)B_ * 2048 * 4);
    float* xibuf  = (float*)take((size_t)B_ * IFPAD_ * 4);
    float* pif    = (float*)take((size_t)B_ * 512 * 4);
    // DNC memory state (contiguous, zeroed in one shot)
    float* mem    = (float*)take((size_t)B_ * N_ * W_ * 4);   // 524288
    float* rw     = (float*)take((size_t)B_ * R_ * N_ * 4);   //  32768
    float* ww     = (float*)take((size_t)B_ * N_ * 4);        //   8192
    float* usage  = (float*)take((size_t)B_ * N_ * 4);        //   8192
    float* link   = (float*)take((size_t)B_ * N_ * N_ * 4);   // 4194304
    float* prec   = (float*)take((size_t)B_ * N_ * 4);        //   8192
    // per-step temporaries
    float* wcw    = (float*)take((size_t)B_ * N_ * 4);
    float* allocw = (float*)take((size_t)B_ * N_ * 4);
    float* cw     = (float*)take((size_t)B_ * R_ * N_ * 4);
    float* fw     = (float*)take((size_t)B_ * R_ * N_ * 4);
    float* bw     = (float*)take((size_t)B_ * R_ * N_ * 4);
    float* rvec   = (float*)take((size_t)B_ * RW_ * 4);

    // ---- per-call prep (deterministic; ws is not preserved across calls) ----
    k_pack_B<<<6144, 256, 0, stream>>>(W_ih0, 512, W_hh0, 512, 256, 768, 2048, 2048, Bp0);
    k_pack_B<<<8192, 256, 0, stream>>>(W_ih1, 512, W_hh1, 512, 512, 1024, 2048, 2048, Bp1);
    k_pack_B<<<960,  256, 0, stream>>>(W_xi,  512, W_xi,  512, 512, 512, IFPAD_, IFACE_, BpXi);
    k_pack_B<<<768,  256, 0, stream>>>(W_out, 768, W_out, 768, 768, 768, 256, 256, BpY);
    k_vecadd <<<8, 256, 0, stream>>>(bg0, b_ih0, b_hh0, 2048);
    k_vecadd <<<8, 256, 0, stream>>>(bg1, b_ih1, b_hh1, 2048);
    k_padcopy<<<2, 256, 0, stream>>>(bxi, b_xi, IFACE_, IFPAD_);
    k_h0init <<<32, 256, 0, stream>>>(h0, ha, ca, hb, cb);
    k_zero   <<<18656, 256, 0, stream>>>(mem, 4775936);  // mem..prec contiguous

    // ---- sequential scan over timesteps ----
    for (int t = 0; t < T_; ++t) {
        const float* xt = x + (size_t)t * IN_;           // row stride T_*IN_

        // LSTM layer 0: gates0 = [x_t | ha] @ W0^T + b   (128 tiles -> 64 wave-pairs)
        k_wmma_gemm<<<16, 128, 0, stream>>>(Bp0, xt, T_ * IN_, ha, H_, 256,
                                            bg0, gates0, 2048, 768, 2048);
        k_lstm_act <<<32, 256, 0, stream>>>(gates0, ca, ha, nullptr);

        // LSTM layer 1: gates1 = [ha | hb] @ W1^T + b ; out = clip(hb)
        k_wmma_gemm<<<16, 128, 0, stream>>>(Bp1, ha, H_, hb, H_, 512,
                                            bg1, gates1, 2048, 1024, 2048);
        k_lstm_act <<<32, 256, 0, stream>>>(gates1, cb, hb, outc);

        // interface vector (30 tiles -> 15 wave-pairs; wave 15 exits)
        k_wmma_gemm<<<4, 128, 0, stream>>>(BpXi, outc, H_, outc, H_, 512,
                                           bxi, xibuf, IFPAD_, 512, IFPAD_);
        k_parse_xi<<<16, 512, 0, stream>>>(xibuf, pif);

        // memory controller
        k_usage   <<<32, 256, 0, stream>>>(usage, ww, rw, pif);
        k_content <<<16, 512, 0, stream>>>(pif, mem, wcw, 1, 260, 0, 324);   // write head
        k_alloc   <<<16, 512, 0, stream>>>(usage, allocw);
        k_ww      <<<32, 256, 0, stream>>>(ww, allocw, wcw, pif);
        k_link    <<<16384, 256, 0, stream>>>(link, ww, prec);               // uses old prec
        k_prec    <<<16, 512, 0, stream>>>(prec, ww);
        k_memupd  <<<2048, 256, 0, stream>>>(mem, ww, pif);
        k_content <<<64, 512, 0, stream>>>(pif, mem, cw, 4, 0, 64, 256);     // read heads
        k_fwbw    <<<64, 512, 0, stream>>>(rw, link, fw, bw);                // old rw, new link
        k_rwupdate<<<128, 256, 0, stream>>>(rw, bw, cw, fw, pif);
        k_readvec <<<64, 64, 0, stream>>>(rw, mem, rvec);

        // output projection straight into d_out[:, t, :]  (16 tiles -> 8 pairs)
        k_wmma_gemm<<<2, 128, 0, stream>>>(BpY, outc, H_, rvec, RW_, 512,
                                           b_out, out + (size_t)t * IN_, T_ * IN_,
                                           768, 256);
    }
}
